// ClassicBiLSTMNER_25451976196118
// MI455X (gfx1250) — matrix operand order repaired and checked by reading, not compiled, not run on hardware
//
#include <hip/hip_runtime.h>
#include <hip/hip_bf16.h>

// Problem dims
#define V_   10000
#define E_   300
#define EP_  320     // E padded to multiple of 32 (WMMA K)
#define H_   256
#define G4_  1024    // 4*H
#define L_   9
#define B_   128
#define T_   512
#define TB_  (T_ * B_)
#define NWG_ 8       // workgroups per direction in the scan

typedef __bf16 bf16t;
typedef __attribute__((ext_vector_type(16))) __bf16 v16bf;
typedef __attribute__((ext_vector_type(8)))  float  v8f;
typedef __attribute__((ext_vector_type(4)))  unsigned int u32x4;

union FragAB {
    v16bf v;
    unsigned short us[16];
    u32x4 q[2];
};

__device__ __forceinline__ unsigned short f2bf(float f) {
    unsigned int x = __float_as_uint(f);
    if ((x & 0x7fffffffu) > 0x7f800000u)             // NaN -> quiet NaN
        return (unsigned short)((x >> 16) | 0x0040u);
    unsigned int r = x + 0x7fffu + ((x >> 16) & 1u); // round-to-nearest-even
    return (unsigned short)(r >> 16);
}

__device__ __forceinline__ float sigm(float x) {
    return 1.0f / (1.0f + __expf(-x));
}

// A-matrix 16x32 bf16 fragment (ISA 7.12.2): lane half selects K {0..7,16..23} vs {8..15,24..31}
__device__ __forceinline__ void load_fragA(FragAB& f, const unsigned short* __restrict__ base,
                                           int ld, int row, int k0, int lane) {
    const unsigned short* p = base + (size_t)row * ld + k0 + ((lane >> 4) << 3);
    f.q[0] = *(const u32x4*)p;          // K local 0..7   (VGPR 0..3)
    f.q[1] = *(const u32x4*)(p + 16);   // K local 16..23 (VGPR 4..7)
}

// B-matrix 32x16 bf16 fragment: lane half selects K {0..7,16..23} vs {8..15,24..31} (same as A)
__device__ __forceinline__ void load_fragB(FragAB& f, const unsigned short* __restrict__ base,
                                           int ld, int row, int k0, int lane) {
    const unsigned short* p = base + (size_t)row * ld + k0 + ((lane >> 4) << 3);
    f.q[0] = *(const u32x4*)p;
    f.q[1] = *(const u32x4*)(p + 16);
}

// ---------------- weight convert / pad to bf16 ----------------
__global__ void k_cvt_pad(const float* __restrict__ src, unsigned short* __restrict__ dst,
                          int rows, int scols, int dcols) {
    int idx = blockIdx.x * blockDim.x + threadIdx.x;
    if (idx >= rows * dcols) return;
    int r = idx / dcols, c = idx - r * dcols;
    float v = (c < scols) ? src[(size_t)r * scols + c] : 0.0f;
    dst[idx] = f2bf(v);
}

// ---------------- embedding gather -> bf16 (T,B,EP) ----------------
__global__ void k_gather(const int* __restrict__ ids, const float* __restrict__ emb,
                         unsigned short* __restrict__ xbf) {
    int idx = blockIdx.x * blockDim.x + threadIdx.x;
    if (idx >= TB_ * EP_) return;
    int tb = idx / EP_, e = idx - tb * EP_;
    int t = tb >> 7, b = tb & (B_ - 1);        // tb = t*B + b
    int id = ids[b * T_ + t];                  // input_ids is (B,T)
    float v = (e < E_) ? emb[(size_t)id * E_ + e] : 0.0f;
    xbf[idx] = f2bf(v);
}

// ---------------- zero barrier state + h-exchange read buffer ----------------
__global__ void k_zero(unsigned int* __restrict__ bar, unsigned short* __restrict__ hx) {
    int idx = blockIdx.x * blockDim.x + threadIdx.x;
    if (idx < 4) bar[idx] = 0;
    if (idx < 2 * B_ * H_) {                   // buffer 0 of both directions
        int d = idx >> 15;                     // / (B_*H_)
        int o = idx & (B_ * H_ - 1);
        hx[(size_t)d * 2 * B_ * H_ + o] = 0;
    }
}

// ---------------- input projection GEMM: xp = x @ Wih^T + b ----------------
__global__ void __launch_bounds__(256)
k_proj(const unsigned short* __restrict__ xbf,
       const unsigned short* __restrict__ wih_f, const unsigned short* __restrict__ wih_b,
       const float* __restrict__ bias_f, const float* __restrict__ bias_b,
       float* __restrict__ xp_f, float* __restrict__ xp_b) {
    const int dir = blockIdx.z;
    const unsigned short* w = dir ? wih_b : wih_f;
    const float* bias = dir ? bias_b : bias_f;
    float* out = dir ? xp_b : xp_f;

    const int wave = threadIdx.x >> 5;
    const int lane = threadIdx.x & 31;
    const int mtile = blockIdx.x;                 // 0..4095
    const int ntile = blockIdx.y * 8 + wave;      // 0..63
    const int mrow = mtile * 16 + (lane & 15);
    const int nrow = ntile * 16 + (lane & 15);

    v8f acc;
    const float bval = bias[nrow];
#pragma unroll
    for (int e = 0; e < 8; ++e) acc[e] = bval;

#pragma unroll
    for (int k0 = 0; k0 < EP_; k0 += 32) {
        FragAB a, bf;
        load_fragA(a, xbf, EP_, mrow, k0, lane);
        load_fragB(bf, w, EP_, nrow, k0, lane);
        acc = __builtin_amdgcn_wmma_f32_16x16x32_bf16(false, a.v, false, bf.v,
                                                      (short)0, acc, false, false);
    }

    const int mbase = mtile * 16 + ((lane >> 4) << 3);
#pragma unroll
    for (int e = 0; e < 8; ++e)
        out[(size_t)(mbase + e) * G4_ + nrow] = acc[e];
}

// ---------------- recurrent scan: 8 persistent WGs per direction ----------------
// Whh B-fragments are register-resident for the whole kernel (32 frags = 256 VGPRs/wave,
// wave32 allows 1024; only 2 waves/SIMD here). Per step: 16 A-frag loads from the 64 KB
// L2-resident h buffer + 64 register-operand WMMAs + LSTM cell + one global barrier.
__global__ void __launch_bounds__(256)
k_scan(const float* __restrict__ xp_f, const float* __restrict__ xp_b,
       const unsigned short* __restrict__ whh_f, const unsigned short* __restrict__ whh_b,
       float* __restrict__ hseq_f, float* __restrict__ hseq_b,
       unsigned short* __restrict__ hx, unsigned int* __restrict__ bar) {
    const int dir = blockIdx.x >> 3;   // 0: fwd, 1: bwd
    const int wg  = blockIdx.x & 7;    // 0..7
    const float* xp = dir ? xp_b : xp_f;
    const unsigned short* whh = dir ? whh_b : whh_f;
    float* hseq = dir ? hseq_b : hseq_f;
    unsigned short* hxd = hx + (size_t)dir * 2 * B_ * H_;
    unsigned int* cnt = bar + dir;
    unsigned int* gen = bar + 2 + dir;

    const int wave = threadIdx.x >> 5;
    const int lane = threadIdx.x & 31;
    const int gw = wg * 8 + wave;      // 0..63 global wave id (this direction)
    const int jt = gw & 15;            // h-column tile (fixed per wave)
    const int mg = gw >> 4;            // 0..3 -> batch tiles {2mg, 2mg+1}
    const int mt0 = mg * 2;
    const int ncol = jt * 16 + (lane & 15);

    // preload the wave's entire Whh operand set into registers (4 quads x 8 k-slices)
    FragAB bw[4][8];
#pragma unroll
    for (int q = 0; q < 4; ++q)
#pragma unroll
        for (int kk = 0; kk < 8; ++kk)
            load_fragB(bw[q][kk], whh, H_, q * H_ + ncol, kk * 32, lane);

    float cfr[2][8];
#pragma unroll
    for (int p = 0; p < 2; ++p)
#pragma unroll
        for (int e = 0; e < 8; ++e) cfr[p][e] = 0.0f;

#pragma unroll 1
    for (int s = 0; s < T_; ++s) {
        const int tidx = dir ? (T_ - 1 - s) : s;
        const float* xpt = xp + (size_t)tidx * B_ * G4_;
        const unsigned short* rbuf = hxd + (size_t)(s & 1) * B_ * H_;
        unsigned short* wbuf = hxd + (size_t)((s + 1) & 1) * B_ * H_;

        // prefetch next step's xp tile region into L2 (global_prefetch_b8)
        if (s + 1 < T_) {
            const int tn = dir ? (T_ - 2 - s) : (s + 1);
            const float* nx = xp + (size_t)tn * B_ * G4_ + (size_t)(mg * 32 + lane) * G4_ + jt * 16;
#pragma unroll
            for (int q = 0; q < 4; ++q)
                __builtin_prefetch(nx + q * H_, 0, 3);
        }

        // accumulators init from xp (i,f,g,o quadrants for 2 batch tiles)
        v8f acc[2][4];
#pragma unroll
        for (int p = 0; p < 2; ++p) {
            const int mbase = (mt0 + p) * 16 + ((lane >> 4) << 3);
#pragma unroll
            for (int q = 0; q < 4; ++q)
#pragma unroll
                for (int e = 0; e < 8; ++e)
                    acc[p][q][e] = xpt[(size_t)(mbase + e) * G4_ + q * H_ + ncol];
        }

        // GEMM: B operands come straight from registers
#pragma unroll
        for (int kk = 0; kk < 8; ++kk) {
#pragma unroll
            for (int p = 0; p < 2; ++p) {
                FragAB a;
                load_fragA(a, rbuf, H_, (mt0 + p) * 16 + (lane & 15), kk * 32, lane);
                acc[p][0] = __builtin_amdgcn_wmma_f32_16x16x32_bf16(false, a.v, false, bw[0][kk].v, (short)0, acc[p][0], false, false);
                acc[p][1] = __builtin_amdgcn_wmma_f32_16x16x32_bf16(false, a.v, false, bw[1][kk].v, (short)0, acc[p][1], false, false);
                acc[p][2] = __builtin_amdgcn_wmma_f32_16x16x32_bf16(false, a.v, false, bw[2][kk].v, (short)0, acc[p][2], false, false);
                acc[p][3] = __builtin_amdgcn_wmma_f32_16x16x32_bf16(false, a.v, false, bw[3][kk].v, (short)0, acc[p][3], false, false);
            }
        }

        // LSTM cell (f32) + publish h
#pragma unroll
        for (int p = 0; p < 2; ++p) {
            const int mbase = (mt0 + p) * 16 + ((lane >> 4) << 3);
#pragma unroll
            for (int e = 0; e < 8; ++e) {
                float ig = sigm(acc[p][0][e]);
                float fg = sigm(acc[p][1][e]);
                float gg = tanhf(acc[p][2][e]);
                float og = sigm(acc[p][3][e]);
                float cc = fg * cfr[p][e] + ig * gg;
                cfr[p][e] = cc;
                float h = og * tanhf(cc);
                wbuf[(mbase + e) * H_ + ncol] = f2bf(h);
                hseq[(size_t)tidx * B_ * H_ + (size_t)(mbase + e) * H_ + ncol] = h;
            }
        }

        // global barrier across the 8 WGs of this direction
        __threadfence();
        __syncthreads();
        if (threadIdx.x == 0) {
            unsigned int old = atomicAdd(cnt, 1u);
            if (old == NWG_ - 1) {
                *(volatile unsigned int*)cnt = 0;   // safe: others gated on gen
                __threadfence();
                atomicAdd(gen, 1u);
            }
            while (__hip_atomic_load(gen, __ATOMIC_ACQUIRE, __HIP_MEMORY_SCOPE_AGENT)
                   < (unsigned int)(s + 1)) {
                __builtin_amdgcn_s_sleep(1);
            }
        }
        __syncthreads();
    }
}

// ---------------- emissions: em = [hf|hb] @ Wc^T + bc ----------------
__global__ void k_emis(const float* __restrict__ hf, const float* __restrict__ hb,
                       const float* __restrict__ Wc, const float* __restrict__ bc,
                       float* __restrict__ em) {
    int idx = blockIdx.x * blockDim.x + threadIdx.x;
    if (idx >= TB_ * L_) return;
    int tb = idx / L_, l = idx - tb * L_;
    const float* wf = Wc + (size_t)l * 2 * H_;
    const float* wb = wf + H_;
    const float* pf = hf + (size_t)tb * H_;
    const float* pb = hb + (size_t)tb * H_;
    float acc = bc[l];
    for (int k = 0; k < H_; ++k) acc += pf[k] * wf[k] + pb[k] * wb[k];
    em[idx] = acc;
}

// ---------------- CRF: gold score + log-partition, -mean(llh) ----------------
__global__ void __launch_bounds__(128)
k_crf(const float* __restrict__ em, const int* __restrict__ labels,
      const int* __restrict__ mask, const float* __restrict__ trans,
      const float* __restrict__ startt, const float* __restrict__ endt,
      float* __restrict__ out) {
    __shared__ float sT[L_ * L_];
    __shared__ float sS[L_], sE[L_];
    __shared__ float red[B_];
    int b = threadIdx.x;
    if (b < L_ * L_) sT[b] = trans[b];
    if (b < L_) { sS[b] = startt[b]; sE[b] = endt[b]; }
    __syncthreads();

    float alpha[L_];
    int tag0 = labels[b * T_];
    float score = sS[tag0] + em[(size_t)b * L_ + tag0];
#pragma unroll
    for (int j = 0; j < L_; ++j) alpha[j] = sS[j] + em[(size_t)b * L_ + j];
    int prev = tag0;
    int seqlen = (mask[b * T_] != 0) ? 1 : 0;

    for (int t = 1; t < T_; ++t) {
        const float* et = em + ((size_t)t * B_ + b) * L_;
        int tg = labels[b * T_ + t];
        int mi = mask[b * T_ + t];
        float m = (mi != 0) ? 1.0f : 0.0f;
        seqlen += (mi != 0);
        score += (sT[prev * L_ + tg] + et[tg]) * m;   // numerator (masked)
        prev = tg;                                    // carry updates unconditionally
        if (mi != 0) {                                // denominator forward step
            float nxt[L_];
#pragma unroll
            for (int j = 0; j < L_; ++j) {
                float mx = -3.4e38f;
#pragma unroll
                for (int i = 0; i < L_; ++i)
                    mx = fmaxf(mx, alpha[i] + sT[i * L_ + j]);
                float ss = 0.0f;
#pragma unroll
                for (int i = 0; i < L_; ++i)
                    ss += __expf(alpha[i] + sT[i * L_ + j] - mx);
                nxt[j] = mx + __logf(ss) + et[j];
            }
#pragma unroll
            for (int j = 0; j < L_; ++j) alpha[j] = nxt[j];
        }
    }
    int last = labels[b * T_ + (seqlen - 1)];
    score += sE[last];
    float mx = -3.4e38f;
#pragma unroll
    for (int j = 0; j < L_; ++j) mx = fmaxf(mx, alpha[j] + sE[j]);
    float ss = 0.0f;
#pragma unroll
    for (int j = 0; j < L_; ++j) ss += __expf(alpha[j] + sE[j] - mx);
    float logZ = mx + __logf(ss);

    red[b] = score - logZ;
    __syncthreads();
    for (int st = 64; st > 0; st >>= 1) {
        if (b < st) red[b] += red[b + st];
        __syncthreads();
    }
    if (b == 0) out[0] = -red[0] / (float)B_;
}

// ---------------- launch ----------------
extern "C" void kernel_launch(void* const* d_in, const int* in_sizes, int n_in,
                              void* d_out, int out_size, void* d_ws, size_t ws_size,
                              hipStream_t stream) {
    (void)in_sizes; (void)n_in; (void)out_size; (void)ws_size;
    const int*   ids    = (const int*)d_in[0];
    const int*   amask  = (const int*)d_in[1];
    const int*   labels = (const int*)d_in[2];
    const float* emb    = (const float*)d_in[3];
    const float* Wih_f  = (const float*)d_in[4];
    const float* Whh_f  = (const float*)d_in[5];
    const float* b_f    = (const float*)d_in[6];
    const float* Wih_b  = (const float*)d_in[7];
    const float* Whh_b  = (const float*)d_in[8];
    const float* b_b    = (const float*)d_in[9];
    const float* Wc     = (const float*)d_in[10];
    const float* bc     = (const float*)d_in[11];
    const float* trans  = (const float*)d_in[12];
    const float* st     = (const float*)d_in[13];
    const float* et     = (const float*)d_in[14];
    float* out = (float*)d_out;

    // workspace carve-out (~718 MB total)
    char* ws = (char*)d_ws;
    size_t off = 0;
    auto carve = [&](size_t bytes) -> void* {
        void* p = ws + off;
        off += (bytes + 255) & ~(size_t)255;
        return p;
    };
    unsigned short* xbf  = (unsigned short*)carve((size_t)TB_ * EP_ * 2);   // 40 MB
    unsigned short* wihf = (unsigned short*)carve((size_t)G4_ * EP_ * 2);
    unsigned short* wihb = (unsigned short*)carve((size_t)G4_ * EP_ * 2);
    unsigned short* whhf = (unsigned short*)carve((size_t)G4_ * H_ * 2);
    unsigned short* whhb = (unsigned short*)carve((size_t)G4_ * H_ * 2);
    float* xpf  = (float*)carve((size_t)TB_ * G4_ * 4);                     // 256 MB
    float* xpb  = (float*)carve((size_t)TB_ * G4_ * 4);                     // 256 MB
    float* hf   = (float*)carve((size_t)TB_ * H_ * 4);                      // 67 MB
    float* hb   = (float*)carve((size_t)TB_ * H_ * 4);                      // 67 MB
    float* em   = (float*)carve((size_t)TB_ * L_ * 4);                      // 2.4 MB
    unsigned short* hx = (unsigned short*)carve((size_t)2 * 2 * B_ * H_ * 2); // 256 KB (2 dirs x 2 bufs)
    unsigned int*  bar = (unsigned int*)carve(4 * sizeof(unsigned int));

    const int THR = 256;
    k_cvt_pad<<<(G4_ * EP_ + THR - 1) / THR, THR, 0, stream>>>(Wih_f, wihf, G4_, E_, EP_);
    k_cvt_pad<<<(G4_ * EP_ + THR - 1) / THR, THR, 0, stream>>>(Wih_b, wihb, G4_, E_, EP_);
    k_cvt_pad<<<(G4_ * H_  + THR - 1) / THR, THR, 0, stream>>>(Whh_f, whhf, G4_, H_, H_);
    k_cvt_pad<<<(G4_ * H_  + THR - 1) / THR, THR, 0, stream>>>(Whh_b, whhb, G4_, H_, H_);
    k_gather<<<((size_t)TB_ * EP_ + THR - 1) / THR, THR, 0, stream>>>(ids, emb, xbf);
    k_zero<<<(2 * B_ * H_ + THR - 1) / THR, THR, 0, stream>>>(bar, hx);

    dim3 pg(TB_ / 16, G4_ / 16 / 8, 2);
    k_proj<<<pg, 256, 0, stream>>>(xbf, wihf, wihb, b_f, b_b, xpf, xpb);

    k_scan<<<2 * NWG_, 256, 0, stream>>>(xpf, xpb, whhf, whhb, hf, hb, hx, bar);

    k_emis<<<(TB_ * L_ + THR - 1) / THR, THR, 0, stream>>>(hf, hb, Wc, bc, em);

    k_crf<<<1, 128, 0, stream>>>(em, labels, amask, trans, st, et, out);
}
